// Net_27161373180324
// MI455X (gfx1250) — 8-bit matrix operand order repaired and checked by reading, compiled, not run on hardware
//
#include <hip/hip_runtime.h>

#define F_IN 512
#define HID  128
#define CLS  64

typedef __attribute__((ext_vector_type(16))) int   v16i;
typedef __attribute__((ext_vector_type(8)))  float v8f;

__device__ __forceinline__ float fsign(float v) {
    return (v > 0.f) ? 1.f : ((v < 0.f) ? -1.f : 0.f);
}

__device__ __forceinline__ unsigned int fp8sign(float v) {
    return (v > 0.f) ? 0x38u : ((v < 0.f) ? 0xB8u : 0x00u);
}

__device__ __forceinline__ v8f zero_v8f() {
    v8f z = {0.f, 0.f, 0.f, 0.f, 0.f, 0.f, 0.f, 0.f};
    return z;
}

__device__ __forceinline__ v16i load_a_frag_fp8(const unsigned char* rowbase, int h) {
    v16i a;
#pragma unroll
    for (int q = 0; q < 4; ++q) {
        uint4 c = *reinterpret_cast<const uint4*>(rowbase + q * 32 + h * 16);
        a[q * 4 + 0] = (int)c.x;
        a[q * 4 + 1] = (int)c.y;
        a[q * 4 + 2] = (int)c.z;
        a[q * 4 + 3] = (int)c.w;
    }
    return a;
}

__device__ __forceinline__ v16i load_b_frag_fp8(const unsigned char* colbase, int h) {
    v16i b;
#pragma unroll
    for (int g = 0; g < 4; ++g) {
        uint4 c = *reinterpret_cast<const uint4*>(colbase + g * 32 + h * 16);
        b[g * 4 + 0] = (int)c.x;
        b[g * 4 + 1] = (int)c.y;
        b[g * 4 + 2] = (int)c.z;
        b[g * 4 + 3] = (int)c.w;
    }
    return b;
}

__global__ void fill_f32(float* __restrict__ p, float v, int n) {
    int i = blockIdx.x * blockDim.x + threadIdx.x;
    if (i < n) p[i] = v;
}

__global__ void __launch_bounds__(512)
bn_stats(const float* __restrict__ x, float* __restrict__ colsum,
         float* __restrict__ colsumsq, int N) {
    int t  = threadIdx.x;
    int nb = gridDim.x;
    int rows_per = (N + nb - 1) / nb;
    int r0 = blockIdx.x * rows_per;
    int r1 = min(N, r0 + rows_per);
    float s = 0.f, ss = 0.f;
    for (int r = r0; r < r1; ++r) {
        float v = x[(size_t)r * F_IN + t];
        s += v; ss += v * v;
    }
    atomicAdd(&colsum[t], s);
    atomicAdd(&colsumsq[t], ss);
}

__global__ void __launch_bounds__(512)
bn_finalize(const float* __restrict__ colsum, const float* __restrict__ colsumsq,
            float* __restrict__ mu, float* __restrict__ rstd, int N) {
    int t = threadIdx.x;
    float m = colsum[t] / (float)N;
    float var = colsumsq[t] / (float)N - m * m;
    mu[t]   = m;
    rstd[t] = rsqrtf(var + 1e-5f);
}

__global__ void deg_count(const int* __restrict__ col, float* __restrict__ deg, int E) {
    int e = blockIdx.x * blockDim.x + threadIdx.x;
    if (e < E) atomicAdd(&deg[col[e]], 1.0f);
}

__global__ void dinv_kernel(const float* __restrict__ deg, float* __restrict__ dinv, int N) {
    int i = blockIdx.x * blockDim.x + threadIdx.x;
    if (i < N) {
        float d = deg[i];
        dinv[i] = (d > 0.f) ? rsqrtf(fmaxf(d, 1.0f)) : 0.f;
    }
}

__global__ void binw_kernel(const float* __restrict__ W, unsigned char* __restrict__ signWt,
                            float* __restrict__ wscale, int K, int M) {
    int j = blockIdx.x * blockDim.x + threadIdx.x;
    if (j >= M) return;
    float s = 0.f;
    for (int k = 0; k < K; ++k) s += fabsf(W[(size_t)k * M + j]);
    wscale[j] = s / (float)K;
    for (int k = 0; k < K; ++k)
        signWt[(size_t)j * K + k] = (unsigned char)fp8sign(W[(size_t)k * M + j]);
}

__global__ void __launch_bounds__(128)
binact1(const float* __restrict__ x, const float* __restrict__ mu,
        const float* __restrict__ rstd, unsigned char* __restrict__ sX,
        float* __restrict__ aScale, int N) {
    int i = blockIdx.x;
    int t = threadIdx.x;
    __shared__ float red[128];
    if (i >= N) {
        reinterpret_cast<unsigned int*>(sX + (size_t)i * F_IN)[t] = 0u;
        if (t == 0) aScale[i] = 0.f;
        return;
    }
    const float4 v = reinterpret_cast<const float4*>(x + (size_t)i * F_IN)[t];
    int f0 = t * 4;
    float n0 = (v.x - mu[f0 + 0]) * rstd[f0 + 0];
    float n1 = (v.y - mu[f0 + 1]) * rstd[f0 + 1];
    float n2 = (v.z - mu[f0 + 2]) * rstd[f0 + 2];
    float n3 = (v.w - mu[f0 + 3]) * rstd[f0 + 3];
    red[t] = fabsf(n0) + fabsf(n1) + fabsf(n2) + fabsf(n3);
    __syncthreads();
    for (int off = 64; off > 0; off >>= 1) {
        if (t < off) red[t] += red[t + off];
        __syncthreads();
    }
    unsigned int packed = fp8sign(n0) | (fp8sign(n1) << 8) |
                          (fp8sign(n2) << 16) | (fp8sign(n3) << 24);
    reinterpret_cast<unsigned int*>(sX + (size_t)i * F_IN)[t] = packed;
    if (t == 0) aScale[i] = red[0] / (float)F_IN;
}

__global__ void __launch_bounds__(128)
binact2(const float* __restrict__ agg, unsigned char* __restrict__ sH,
        float* __restrict__ aScale, int N) {
    int i = blockIdx.x;
    int t = threadIdx.x;
    __shared__ float red[128];
    if (i >= N) {
        sH[(size_t)i * HID + t] = 0;
        if (t == 0) aScale[i] = 0.f;
        return;
    }
    float v = agg[(size_t)i * HID + t];
    red[t] = fabsf(v);
    __syncthreads();
    for (int off = 64; off > 0; off >>= 1) {
        if (t < off) red[t] += red[t + off];
        __syncthreads();
    }
    sH[(size_t)i * HID + t] = (unsigned char)fp8sign(v);
    if (t == 0) aScale[i] = red[0] / (float)HID;
}

__global__ void __launch_bounds__(256)
gemm1_wmma(const unsigned char* __restrict__ sX, const unsigned char* __restrict__ sWt,
           const float* __restrict__ aScale, const float* __restrict__ wScale,
           float* __restrict__ h1) {
    const int KC = 128;
    __shared__ unsigned char sW[HID * KC];
    int lane = threadIdx.x & 31;
    int wid  = threadIdx.x >> 5;
    int m0   = blockIdx.x * 128 + wid * 16;
    int h    = lane >> 4;
    int nb   = lane & 15;
    v8f acc[8];
#pragma unroll
    for (int t = 0; t < 8; ++t) acc[t] = zero_v8f();

    for (int k0 = 0; k0 < F_IN; k0 += KC) {
        __syncthreads();
        for (int u = threadIdx.x; u < (HID * KC) / 16; u += 256) {
            int n  = (u * 16) / KC;
            int kk = (u * 16) % KC;
            *reinterpret_cast<uint4*>(&sW[n * KC + kk]) =
                *reinterpret_cast<const uint4*>(&sWt[(size_t)n * F_IN + k0 + kk]);
        }
        __syncthreads();

        v16i a = load_a_frag_fp8(sX + (size_t)(m0 + nb) * F_IN + k0, h);
#pragma unroll
        for (int g2 = 0; g2 < 2; ++g2) {
            v16i b[4];
#pragma unroll
            for (int t = 0; t < 4; ++t)
                b[t] = load_b_frag_fp8(sW + (size_t)((g2 * 4 + t) * 16 + nb) * KC, h);
#pragma unroll
            for (int t = 0; t < 4; ++t)
                acc[g2 * 4 + t] = __builtin_amdgcn_wmma_f32_16x16x128_fp8_fp8(
                    a, b[t], (short)0, acc[g2 * 4 + t], false, false);
        }
    }
#pragma unroll
    for (int t = 0; t < 8; ++t) {
        int n = t * 16 + nb;
        float wsc = wScale[n];
#pragma unroll
        for (int r = 0; r < 8; ++r) {
            int m = m0 + 8 * h + r;
            h1[(size_t)m * HID + n] = acc[t][r] * aScale[m] * wsc;
        }
    }
}

__global__ void __launch_bounds__(256)
gemm2_wmma(const unsigned char* __restrict__ sH, const unsigned char* __restrict__ sWt,
           const float* __restrict__ aScale, const float* __restrict__ wScale,
           float* __restrict__ h2) {
    __shared__ unsigned char sW[CLS * HID];
    for (int u = threadIdx.x; u < (CLS * HID) / 16; u += 256)
        *reinterpret_cast<uint4*>(&sW[u * 16]) =
            *reinterpret_cast<const uint4*>(&sWt[u * 16]);
    __syncthreads();

    int lane = threadIdx.x & 31;
    int wid  = threadIdx.x >> 5;
    int m0   = blockIdx.x * 128 + wid * 16;
    int h    = lane >> 4;
    int nb   = lane & 15;

    v16i a = load_a_frag_fp8(sH + (size_t)(m0 + nb) * HID, h);
    v16i b[4];
#pragma unroll
    for (int t = 0; t < 4; ++t)
        b[t] = load_b_frag_fp8(sW + (size_t)(t * 16 + nb) * HID, h);

    v8f acc[4];
#pragma unroll
    for (int t = 0; t < 4; ++t) {
        acc[t] = zero_v8f();
        acc[t] = __builtin_amdgcn_wmma_f32_16x16x128_fp8_fp8(
            a, b[t], (short)0, acc[t], false, false);
    }
#pragma unroll
    for (int t = 0; t < 4; ++t) {
        int n = t * 16 + nb;
        float wsc = wScale[n];
#pragma unroll
        for (int r = 0; r < 8; ++r) {
            int m = m0 + 8 * h + r;
            h2[(size_t)m * CLS + n] = acc[t][r] * aScale[m] * wsc;
        }
    }
}

__global__ void scatter_init(const float* __restrict__ hmat, const float* __restrict__ dinv,
                             const float* __restrict__ bias, float* __restrict__ agg,
                             int N, int Npad, int F) {
    size_t idx = (size_t)blockIdx.x * blockDim.x + threadIdx.x;
    if (idx >= (size_t)Npad * F) return;
    int i = (int)(idx / F);
    int f = (int)(idx % F);
    if (i < N) {
        float d = dinv[i];
        agg[idx] = bias[f] + d * d * hmat[idx];
    } else {
        agg[idx] = 0.f;
    }
}

__global__ void __launch_bounds__(256)
scatter_edges(const int* __restrict__ row, const int* __restrict__ col,
              const float* __restrict__ dinv, const float* __restrict__ hmat,
              float* __restrict__ agg, int E, int F) {
    int epb = 256 / F;
    int e = blockIdx.x * epb + threadIdx.x / F;
    int f = threadIdx.x % F;
    if (e >= E) return;
    int r = row[e], c = col[e];
    float w = dinv[r] * dinv[c];
    atomicAdd(&agg[(size_t)c * F + f], w * hmat[(size_t)r * F + f]);
}

__global__ void __launch_bounds__(64)
logsoftmax64(const float* __restrict__ agg, float* __restrict__ out) {
    int i = blockIdx.x;
    int t = threadIdx.x;
    __shared__ float red[64];
    float z = agg[(size_t)i * CLS + t];
    red[t] = z;
    __syncthreads();
    for (int off = 32; off > 0; off >>= 1) {
        if (t < off) red[t] = fmaxf(red[t], red[t + off]);
        __syncthreads();
    }
    float mx = red[0];
    __syncthreads();
    red[t] = expf(z - mx);
    __syncthreads();
    for (int off = 32; off > 0; off >>= 1) {
        if (t < off) red[t] += red[t + off];
        __syncthreads();
    }
    float lse = mx + logf(red[0]);
    out[(size_t)i * CLS + t] = z - lse;
}

extern "C" void kernel_launch(void* const* d_in, const int* in_sizes, int n_in,
                              void* d_out, int out_size, void* d_ws, size_t ws_size,
                              hipStream_t stream) {
    const float* x    = (const float*)d_in[0];
    const int*   edge = (const int*)d_in[1];
    const float* W1   = (const float*)d_in[2];
    const float* b1   = (const float*)d_in[3];
    const float* W2   = (const float*)d_in[4];
    const float* b2   = (const float*)d_in[5];
    float* out = (float*)d_out;

    const int N    = in_sizes[0] / F_IN;
    const int E    = in_sizes[1] / 2;
    const int Npad = ((N + 127) / 128) * 128;
    const int* row = edge;
    const int* col = edge + E;

    char* wsb = (char*)d_ws;
    size_t off = 0;
    auto alloc = [&](size_t bytes) -> void* {
        void* p = wsb + off;
        off += bytes;
        off = (off + 255) & ~(size_t)255;
        return p;
    };
    float* colsum   = (float*)alloc(F_IN * 4);
    float* colsumsq = (float*)alloc(F_IN * 4);
    float* mu       = (float*)alloc(F_IN * 4);
    float* rstd     = (float*)alloc(F_IN * 4);
    float* deg      = (float*)alloc((size_t)N * 4);
    float* dinv     = (float*)alloc((size_t)N * 4);
    float* aS1      = (float*)alloc((size_t)Npad * 4);
    float* aS2      = (float*)alloc((size_t)Npad * 4);
    float* wS1      = (float*)alloc(HID * 4);
    float* wS2      = (float*)alloc(CLS * 4);
    unsigned char* sX   = (unsigned char*)alloc((size_t)Npad * F_IN);
    unsigned char* sW1t = (unsigned char*)alloc((size_t)HID * F_IN);
    unsigned char* sH   = (unsigned char*)alloc((size_t)Npad * HID);
    unsigned char* sW2t = (unsigned char*)alloc((size_t)CLS * HID);
    float* h1       = (float*)alloc((size_t)Npad * HID * 4);
    float* h2       = (float*)alloc((size_t)Npad * CLS * 4);
    float* agg1     = (float*)alloc((size_t)Npad * HID * 4);
    float* agg2     = (float*)alloc((size_t)Npad * CLS * 4);

    fill_f32<<<(F_IN + 255) / 256, 256, 0, stream>>>(colsum, 0.f, F_IN);
    fill_f32<<<(F_IN + 255) / 256, 256, 0, stream>>>(colsumsq, 0.f, F_IN);
    bn_stats<<<128, 512, 0, stream>>>(x, colsum, colsumsq, N);
    bn_finalize<<<1, 512, 0, stream>>>(colsum, colsumsq, mu, rstd, N);

    fill_f32<<<(N + 255) / 256, 256, 0, stream>>>(deg, 1.0f, N);
    deg_count<<<(E + 255) / 256, 256, 0, stream>>>(col, deg, E);
    dinv_kernel<<<(N + 255) / 256, 256, 0, stream>>>(deg, dinv, N);

    binw_kernel<<<1, HID, 0, stream>>>(W1, sW1t, wS1, F_IN, HID);
    binw_kernel<<<1, CLS, 0, stream>>>(W2, sW2t, wS2, HID, CLS);

    binact1<<<Npad, 128, 0, stream>>>(x, mu, rstd, sX, aS1, N);
    gemm1_wmma<<<Npad / 128, 256, 0, stream>>>(sX, sW1t, aS1, wS1, h1);
    {
        size_t tot = (size_t)Npad * HID;
        scatter_init<<<(unsigned)((tot + 255) / 256), 256, 0, stream>>>(
            h1, dinv, b1, agg1, N, Npad, HID);
        scatter_edges<<<(E + 1) / 2, 256, 0, stream>>>(row, col, dinv, h1, agg1, E, HID);
    }

    binact2<<<Npad, 128, 0, stream>>>(agg1, sH, aS2, N);
    gemm2_wmma<<<Npad / 128, 256, 0, stream>>>(sH, sW2t, aS2, wS2, h2);
    {
        size_t tot = (size_t)Npad * CLS;
        scatter_init<<<(unsigned)((tot + 255) / 256), 256, 0, stream>>>(
            h2, dinv, b2, agg2, N, Npad, CLS);
        scatter_edges<<<(E + 3) / 4, 256, 0, stream>>>(row, col, dinv, h2, agg2, E, CLS);
    }

    logsoftmax64<<<N, 64, 0, stream>>>(agg2, out);
}
